// DHEL_57647051047405
// MI455X (gfx1250) — matrix operand order repaired and checked by reading, not compiled, not run on hardware
//
#include <hip/hip_runtime.h>
#include <hip/hip_bf16.h>
#include <cstdint>

// ---------------------------------------------------------------------------
// DHEL contrastive loss, MI455X / gfx1250.
//   z: 16384 x 128 fp32.  m = 8192.  tau = 0.3.
//   loss = mean_i[ log S_a(i) + log S_p(i) - dot(za_i, zp_i)/tau ]
//   S_x(i) = sum_{j != i} exp( (x_i . x_j)/tau )  on row-normalized vectors.
// Compute-bound (34 GFLOP GEMM vs 8.4 MB data, fully L2-resident) ->
// f16 WMMA w/ fp32 accumulation, row sums fused into the GEMM epilogue.
// B tiles: double-buffered async global->LDS DMA overlapped with WMMA.
// ---------------------------------------------------------------------------

typedef _Float16 v16h __attribute__((ext_vector_type(16)));
typedef _Float16 h8   __attribute__((ext_vector_type(8)));
typedef _Float16 h4   __attribute__((ext_vector_type(4)));
typedef float    v8f  __attribute__((ext_vector_type(8)));
typedef int      v4i  __attribute__((ext_vector_type(4)));
typedef __attribute__((address_space(1))) v4i* g_v4i_p;
typedef __attribute__((address_space(3))) v4i* l_v4i_p;

#define NROWS 16384
#define MHALF 8192
#define DDIM  128
#define BN    32                                  // B columns staged per iter
#define XSPLIT 4                                  // column splits of the sweep
#define TILES_PER_SPLIT (MHALF / BN / XSPLIT)     // 64
#define LDSROW 136                                // 128 halves + 8 pad
static __device__ __forceinline__ float inv_tau() { return 1.0f / 0.3f; }

#if defined(__has_builtin)
#if __has_builtin(__builtin_amdgcn_global_load_async_to_lds_b128)
#define HAVE_ASYNC_LDS 1
#endif
#endif

static __device__ __forceinline__ void lds_fill16(const _Float16* src, _Float16* dst) {
#ifdef HAVE_ASYNC_LDS
  __builtin_amdgcn_global_load_async_to_lds_b128(
      (g_v4i_p)(uintptr_t)src, (l_v4i_p)(uintptr_t)dst, 0, 0);
  __builtin_amdgcn_global_load_async_to_lds_b128(
      (g_v4i_p)(uintptr_t)(src + 8), (l_v4i_p)(uintptr_t)(dst + 8), 0, 0);
#else
  *(h8*)dst = *(const h8*)src;
  *(h8*)(dst + 8) = *(const h8*)(src + 8);
#endif
}

static __device__ __forceinline__ void wait_async_keep2() {
#ifdef HAVE_ASYNC_LDS
#if defined(__has_builtin) && __has_builtin(__builtin_amdgcn_s_wait_asynccnt)
  __builtin_amdgcn_s_wait_asynccnt(2);
#else
  asm volatile("s_wait_asynccnt 2" ::: "memory");
#endif
#endif
}
static __device__ __forceinline__ void wait_async_all() {
#ifdef HAVE_ASYNC_LDS
#if defined(__has_builtin) && __has_builtin(__builtin_amdgcn_s_wait_asynccnt)
  __builtin_amdgcn_s_wait_asynccnt(0);
#else
  asm volatile("s_wait_asynccnt 0" ::: "memory");
#endif
#endif
}

// --------------------------- kernel 1: normalize ---------------------------
// one wave (32 lanes) per row; 4 floats per lane; store f16 normalized row.
__global__ __launch_bounds__(256) void dhel_normalize(const float* __restrict__ z,
                                                      _Float16* __restrict__ zh) {
  const int row  = blockIdx.x * 8 + (threadIdx.x >> 5);
  const int lane = threadIdx.x & 31;
  const float4 v = ((const float4*)(z + (size_t)row * DDIM))[lane];
  float ss = v.x * v.x + v.y * v.y + v.z * v.z + v.w * v.w;
  ss += __shfl_xor(ss, 16);
  ss += __shfl_xor(ss, 8);
  ss += __shfl_xor(ss, 4);
  ss += __shfl_xor(ss, 2);
  ss += __shfl_xor(ss, 1);
  const float inv = 1.0f / fmaxf(sqrtf(ss), 1e-12f);
  h4 h;
  h.x = (_Float16)(v.x * inv);
  h.y = (_Float16)(v.y * inv);
  h.z = (_Float16)(v.z * inv);
  h.w = (_Float16)(v.w * inv);
  *(h4*)(zh + (size_t)row * DDIM + lane * 4) = h;
}

// --------------------------- kernel 2: WMMA row sums -----------------------
// grid: (64 row-blocks, XSPLIT col-splits, 2 matrix halves), 256 threads.
// Each wave owns a 16-row A stripe (resident in VGPRs, K=128 = 4 wmma chunks)
// and computes two 16x16 tiles per iteration against a 32-column B tile that
// is double-buffered in LDS via async DMA (fill of tile j+1 overlaps WMMA j).
__global__ __launch_bounds__(256) void dhel_rowsum(const _Float16* __restrict__ zh,
                                                   float* __restrict__ Spart) {
  const int half  = blockIdx.z;                 // 0 = za, 1 = zp
  const int split = blockIdx.y;
  const _Float16* zb = zh + (size_t)half * MHALF * DDIM;

  __shared__ _Float16 Bs[2][BN * LDSROW];       // 2 x 8.5 KB ping-pong

  const int tid  = threadIdx.x;
  const int wave = tid >> 5;
  const int lane = tid & 31;
  const int g    = lane >> 4;                   // K-group within wave
  const int ln   = lane & 15;

  const int rbase = blockIdx.x * 128 + wave * 16;

  // ---- load A stripe: rows rbase..rbase+15, ISA 16x32-f16 A layout ----
  // lane group g: halves 0..7 -> k = 32c + 8g + h ; halves 8..15 -> 32c+16+8g+(h-8)
  v16h a[4];
  {
    const _Float16* arow = zb + (size_t)(rbase + ln) * DDIM;
#pragma unroll
    for (int c = 0; c < 4; ++c) {
      union { v16h v; h8 h[2]; } u;
      u.h[0] = *(const h8*)(arow + 32 * c + 8 * g);
      u.h[1] = *(const h8*)(arow + 32 * c + 16 + 8 * g);
      a[c] = u.v;
    }
  }

  float acc[8];
#pragma unroll
  for (int v = 0; v < 8; ++v) acc[v] = 0.0f;

  // cooperative fill geometry: 256 threads x 32 bytes = 32 rows x 256 B
  const int frow = tid >> 3;                    // 0..31
  const int fseg = tid & 7;                     // 0..7, 16 halves each

  const int t0 = split * TILES_PER_SPLIT;

  // prologue: fill buffer 0 with tile t0
  lds_fill16(zb + (size_t)(t0 * BN + frow) * DDIM + fseg * 16,
             &Bs[0][frow * LDSROW + fseg * 16]);

  for (int it = 0; it < TILES_PER_SPLIT; ++it) {
    const int buf = it & 1;
    const int cbase = (t0 + it) * BN;

    // issue fill of the next tile into the other buffer (overlaps WMMA below)
    if (it + 1 < TILES_PER_SPLIT) {
      lds_fill16(zb + (size_t)((t0 + it + 1) * BN + frow) * DDIM + fseg * 16,
                 &Bs[buf ^ 1][frow * LDSROW + fseg * 16]);
      wait_async_keep2();                       // current fill done; next in flight
    } else {
      wait_async_all();
    }
    __syncthreads();

    // ---- two 16x16 tiles: 8 chained v_wmma_f32_16x16x32_f16 over K=128 ----
    v8f ct0 = {}, ct1 = {};
#pragma unroll
    for (int c = 0; c < 4; ++c) {
      // B layout (32x16 f16), same K order as A: lane group g holds
      // K=8g..8g+7 and K=16+8g..16+8g+7 of column N=lane%16.
      union { v16h v; h8 h[2]; } u0, u1;
      const _Float16* b0 = &Bs[buf][ln * LDSROW + 32 * c + 8 * g];
      const _Float16* b1 = &Bs[buf][(16 + ln) * LDSROW + 32 * c + 8 * g];
      u0.h[0] = *(const h8*)(b0);
      u0.h[1] = *(const h8*)(b0 + 16);
      u1.h[0] = *(const h8*)(b1);
      u1.h[1] = *(const h8*)(b1 + 16);
      ct0 = __builtin_amdgcn_wmma_f32_16x16x32_f16(false, a[c], false, u0.v,
                                                   (short)0, ct0, false, false);
      ct1 = __builtin_amdgcn_wmma_f32_16x16x32_f16(false, a[c], false, u1.v,
                                                   (short)0, ct1, false, false);
    }

    // ---- epilogue: exp(c/tau), diagonal masked, per-row accumulate ----
    // C layout: lanes 0-15 -> N=lane, M=v ; lanes 16-31 -> N=lane-16, M=v+8.
#pragma unroll
    for (int v = 0; v < 8; ++v) {
      const int grow = rbase + 8 * g + v;
      float e0 = __expf(ct0[v] * inv_tau());
      float e1 = __expf(ct1[v] * inv_tau());
      if (grow == cbase + ln) e0 = 0.0f;        // exclude j == i
      if (grow == cbase + 16 + ln) e1 = 0.0f;
      acc[v] += e0 + e1;
    }
    __syncthreads();                            // done reading Bs[buf]
  }

  // reduce each row's partial over the 16 lanes that hold its columns
#pragma unroll
  for (int v = 0; v < 8; ++v) {
    float s = acc[v];
    s += __shfl_xor(s, 1);
    s += __shfl_xor(s, 2);
    s += __shfl_xor(s, 4);
    s += __shfl_xor(s, 8);
    acc[v] = s;
  }
  if (ln == 0) {
    float* Sb = Spart + (size_t)(half * XSPLIT + split) * MHALF;
#pragma unroll
    for (int v = 0; v < 8; ++v) Sb[rbase + 8 * g + v] = acc[v];
  }
}

// --------------------------- kernel 3: per-row loss terms ------------------
// one wave per row i: recompute exact fp32 normalized dot(za_i, zp_i),
// combine with row sums; block-reduce into partial[blockIdx] (deterministic).
__global__ __launch_bounds__(256) void dhel_loss1(const float* __restrict__ z,
                                                  const float* __restrict__ Spart,
                                                  float* __restrict__ partial) {
  __shared__ float wsum[8];
  const int i    = blockIdx.x * 8 + (threadIdx.x >> 5);
  const int lane = threadIdx.x & 31;

  const float4 x = ((const float4*)(z + (size_t)i * DDIM))[lane];
  const float4 y = ((const float4*)(z + (size_t)(i + MHALF) * DDIM))[lane];
  float ssa = x.x * x.x + x.y * x.y + x.z * x.z + x.w * x.w;
  float ssp = y.x * y.x + y.y * y.y + y.z * y.z + y.w * y.w;
  float dp  = x.x * y.x + x.y * y.y + x.z * y.z + x.w * y.w;
#pragma unroll
  for (int m = 16; m >= 1; m >>= 1) {
    ssa += __shfl_xor(ssa, m);
    ssp += __shfl_xor(ssp, m);
    dp  += __shfl_xor(dp, m);
  }
  if (lane == 0) {
    const float na = fmaxf(sqrtf(ssa), 1e-12f);
    const float np = fmaxf(sqrtf(ssp), 1e-12f);
    const float dot = dp / (na * np);
    float Sa = 0.0f, Sp = 0.0f;
#pragma unroll
    for (int s = 0; s < XSPLIT; ++s) {
      Sa += Spart[(size_t)s * MHALF + i];
      Sp += Spart[(size_t)(XSPLIT + s) * MHALF + i];
    }
    wsum[threadIdx.x >> 5] = logf(Sa) + logf(Sp) - dot * inv_tau();
  }
  __syncthreads();
  if (threadIdx.x == 0) {
    float t = 0.0f;
#pragma unroll
    for (int w = 0; w < 8; ++w) t += wsum[w];
    partial[blockIdx.x] = t;
  }
}

// --------------------------- kernel 4: final reduce ------------------------
__global__ __launch_bounds__(256) void dhel_loss2(const float* __restrict__ partial,
                                                  float* __restrict__ out) {
  __shared__ float red[256];
  const int t = threadIdx.x;
  float s = partial[t] + partial[t + 256] + partial[t + 512] + partial[t + 768];
  red[t] = s;
  __syncthreads();
#pragma unroll
  for (int off = 128; off >= 1; off >>= 1) {
    if (t < off) red[t] += red[t + off];
    __syncthreads();
  }
  if (t == 0) out[0] = red[0] * (1.0f / (float)MHALF);
}

// ---------------------------------------------------------------------------
extern "C" void kernel_launch(void* const* d_in, const int* in_sizes, int n_in,
                              void* d_out, int out_size, void* d_ws, size_t ws_size,
                              hipStream_t stream) {
  (void)in_sizes; (void)n_in; (void)out_size; (void)ws_size;
  const float* z = (const float*)d_in[0];
  float* out = (float*)d_out;

  // workspace layout
  float*    Spart   = (float*)d_ws;                                   // 2*XSPLIT*8192 f32 = 256 KB
  float*    partial = (float*)((char*)d_ws + 2 * XSPLIT * MHALF * 4); // 1024 f32
  _Float16* zh      = (_Float16*)((char*)d_ws + 2 * XSPLIT * MHALF * 4 + 4096); // 4 MB

  dhel_normalize<<<NROWS / 8, 256, 0, stream>>>(z, zh);
  dim3 grid(MHALF / 128, XSPLIT, 2);
  dhel_rowsum<<<grid, 256, 0, stream>>>(zh, Spart);
  dhel_loss1<<<MHALF / 8, 256, 0, stream>>>(z, Spart, partial);
  dhel_loss2<<<1, 256, 0, stream>>>(partial, out);
}
